// LongformerSelfAttention_77489799955228
// MI455X (gfx1250) — matrix operand order repaired and checked by reading, not compiled, not run on hardware
//
#include <hip/hip_runtime.h>
#include <hip/hip_bf16.h>

#define BB   2
#define SS   2048
#define HID  768
#define NH   12
#define HD   64
#define WIN  256

typedef __attribute__((ext_vector_type(8)))  float  f32x8;
typedef __attribute__((ext_vector_type(16))) float  f32x16;
typedef __attribute__((ext_vector_type(8)))  __bf16 bf16x8;
typedef __attribute__((ext_vector_type(16))) __bf16 bf16x16;

__device__ __forceinline__ f32x8 wmma_bf16(bf16x16 a, bf16x16 b, f32x8 c) {
    // D(16x16,f32) = A(16x32,bf16) x B(32x16,bf16) + C
    return __builtin_amdgcn_wmma_f32_16x16x32_bf16(
        /*neg_a=*/false, a, /*neg_b=*/false, b,
        /*c_mod=*/(short)0, c, /*reuse_a=*/false, /*reuse_b=*/false);
}

__device__ __forceinline__ int iclamp(int v, int lo, int hi) {
    return v < lo ? lo : (v > hi ? hi : v);
}

// ---------------------------------------------------------------------------
// Kernel 1: QKV projection. grid = (384, 3); block = 256 (8 waves).
// Each wave computes a 16(row) x 64(col = one head) tile of y = x*W^T + b,
// writes bf16 to q / k (layout [B,H,S,D]) or vt (layout [B,H,D,S]).
// ---------------------------------------------------------------------------
__global__ __launch_bounds__(256) void qkv_kernel(
    const float* __restrict__ x,
    const float* __restrict__ Wq, const float* __restrict__ bq,
    const float* __restrict__ Wk, const float* __restrict__ bk,
    const float* __restrict__ Wv, const float* __restrict__ bv,
    __bf16* __restrict__ qd, __bf16* __restrict__ kd, __bf16* __restrict__ vtd)
{
    const int m = blockIdx.y;                       // 0=Q 1=K 2=V
    const float* W  = (m == 0) ? Wq : ((m == 1) ? Wk : Wv);
    const float* bs = (m == 0) ? bq : ((m == 1) ? bk : bv);
    const float scale = (m == 0) ? 0.125f : 1.0f;   // q /= sqrt(64)

    const int lane = threadIdx.x & 31;
    const int wid  = threadIdx.x >> 5;
    const int job  = blockIdx.x * 8 + wid;          // 3072 jobs
    const int rowTile = job & 255;                  // 256 tiles of 16 rows
    const int headT   = job >> 8;                   // 0..11
    const int t0 = rowTile * 16;
    const int hh = lane >> 4, mm = lane & 15;

    const float* xrow = x + (size_t)(t0 + mm) * HID;

    f32x8 acc[4];
    #pragma unroll
    for (int s4 = 0; s4 < 4; s4++)
        acc[s4] = (f32x8){0.f,0.f,0.f,0.f,0.f,0.f,0.f,0.f};

    for (int kk = 0; kk < HID; kk += 32) {
        // A fragment: x[t0+mm][kk + 16hh .. 16hh+15] (contiguous)
        f32x8 r0 = *(const f32x8*)(xrow + kk + 16 * hh);
        f32x8 r1 = *(const f32x8*)(xrow + kk + 16 * hh + 8);
        bf16x16 a;
        #pragma unroll
        for (int e = 0; e < 8; e++) { a[e] = (__bf16)r0[e]; a[e + 8] = (__bf16)r1[e]; }

        #pragma unroll
        for (int s4 = 0; s4 < 4; s4++) {
            // B fragment: B[K,N] = W[n][kk+K]; lane: n = col, 16 contiguous K
            const int n = headT * 64 + s4 * 16 + mm;
            f32x16 wf = *(const f32x16*)(W + (size_t)n * HID + kk + 16 * hh);
            bf16x16 bfr;
            #pragma unroll
            for (int e = 0; e < 16; e++) bfr[e] = (__bf16)wf[e];
            acc[s4] = wmma_bf16(a, bfr, acc[s4]);
        }
    }

    #pragma unroll
    for (int s4 = 0; s4 < 4; s4++) {
        const int d = s4 * 16 + mm;
        const float bias = bs[headT * 64 + d];
        #pragma unroll
        for (int r = 0; r < 8; r++) {
            const int t = t0 + r + 8 * hh;          // global token
            const int bi = t >> 11, s = t & (SS - 1);
            const float y = (acc[s4][r] + bias) * scale;
            const __bf16 yb = (__bf16)y;
            if (m == 2) {
                vtd[((size_t)(bi * NH + headT) * HD + d) * SS + s] = yb;
            } else {
                const size_t o = ((size_t)(bi * NH + headT) * SS + s) * HD + d;
                if (m == 0) qd[o] = yb; else kd[o] = yb;
            }
        }
    }
}

// ---------------------------------------------------------------------------
// Kernel 2: per-(b,h) column sums of V (rows of Vt). grid = 24, block = 64.
// ---------------------------------------------------------------------------
__global__ __launch_bounds__(64) void vsum_kernel(
    const __bf16* __restrict__ vtd, float* __restrict__ vsum)
{
    const int bh = blockIdx.x, d = threadIdx.x;
    const bf16x8* row = (const bf16x8*)(vtd + ((size_t)bh * HD + d) * SS);
    float acc = 0.f;
    for (int i = 0; i < SS / 8; i++) {
        bf16x8 v = row[i];
        #pragma unroll
        for (int e = 0; e < 8; e++) acc += (float)v[e];
    }
    vsum[bh * HD + d] = acc;
}

// ---------------------------------------------------------------------------
// Kernel 3: windowed attention. grid = (16, 12, 2); block = 256 (8 waves).
// Each wave owns 16 queries of one (b,h). Scores computed transposed
// (St = K_tile x Q) so exp-probability fragments land directly in the
// A-matrix layout of the P*V WMMA — no LDS transpose.
// ---------------------------------------------------------------------------
__global__ __launch_bounds__(256) void attn_kernel(
    const __bf16* __restrict__ qd, const __bf16* __restrict__ kd,
    const __bf16* __restrict__ vtd, const float* __restrict__ vsum,
    float* __restrict__ out)
{
    const int lane = threadIdx.x & 31;
    const int wid  = threadIdx.x >> 5;
    const int b = blockIdx.z, h = blockIdx.y;
    const int i0 = blockIdx.x * 128 + wid * 16;
    const int hh = lane >> 4, mm = lane & 15;
    const int bh = b * NH + h;

    const __bf16* qb = qd + (size_t)bh * SS * HD;
    const __bf16* kb = kd + (size_t)bh * SS * HD;
    const __bf16* vb = vtd + (size_t)bh * HD * SS;

    // Q as B-matrix of St: b[e] = q[i0+mm][d0 + 16*hh + e] (contiguous)
    const __bf16* qrow = qb + (size_t)(i0 + mm) * HD;
    const bf16x16 bq0 = *(const bf16x16*)(qrow + 16 * hh);
    const bf16x16 bq1 = *(const bf16x16*)(qrow + 32 + 16 * hh);

    f32x8 ctx[4];
    #pragma unroll
    for (int s4 = 0; s4 < 4; s4++)
        ctx[s4] = (f32x8){0.f,0.f,0.f,0.f,0.f,0.f,0.f,0.f};
    float rs = 0.f;                                  // sum_j (e^s - 1) for i = mm

    for (int jb = i0 - WIN; jb <= i0 + WIN; jb += 32) {
        float p0[8], p1[8];
        #pragma unroll
        for (int t = 0; t < 2; t++) {
            const int jt = jb + 16 * t;
            const int jcl = iclamp(jt, 0, SS - 16);  // 16-aligned, never straddles
            // K tile as A-matrix: row j = jcl+mm, two 8-elem runs per 32-d block
            const __bf16* krow = kb + (size_t)(jcl + mm) * HD;
            bf16x8 a00 = *(const bf16x8*)(krow + 16 * hh);
            bf16x8 a01 = *(const bf16x8*)(krow + 16 * hh + 8);
            bf16x8 a10 = *(const bf16x8*)(krow + 32 + 16 * hh);
            bf16x8 a11 = *(const bf16x8*)(krow + 40 + 16 * hh);
            bf16x16 aA = __builtin_shufflevector(a00, a01,
                0,1,2,3,4,5,6,7,8,9,10,11,12,13,14,15);
            bf16x16 aB = __builtin_shufflevector(a10, a11,
                0,1,2,3,4,5,6,7,8,9,10,11,12,13,14,15);
            f32x8 st = (f32x8){0.f,0.f,0.f,0.f,0.f,0.f,0.f,0.f};
            st = wmma_bf16(aA, bq0, st);             // d = 0..31
            st = wmma_bf16(aB, bq1, st);             // d = 32..63
            // St C-layout: element r -> (j = jt + r + 8hh, i = i0 + mm)
            float* pp = t ? p1 : p0;
            #pragma unroll
            for (int r = 0; r < 8; r++) {
                const int j = jt + r + 8 * hh;
                const int i = i0 + mm;
                const bool ok = (j >= 0) && (j < SS) &&
                                (j - i <= WIN) && (i - j <= WIN);
                const float pe = ok ? (__expf(st[r]) - 1.0f) : 0.0f;
                pp[r] = pe;
                rs += pe;
            }
        }
        // Repack: St C-fragments ARE the A-fragment for P(16x32) x V(32x64)
        bf16x16 ap;
        #pragma unroll
        for (int e = 0; e < 8; e++) { ap[e] = (__bf16)p0[e]; ap[e + 8] = (__bf16)p1[e]; }

        const int jrun0 = iclamp(jb + 8 * hh, 0, SS - 8);
        const int jrun1 = iclamp(jb + 16 + 8 * hh, 0, SS - 8);
        #pragma unroll
        for (int s4 = 0; s4 < 4; s4++) {
            // Vt row d = s4*16+mm, keys jb+8hh..+7 and jb+16+8hh..+7
            const bf16x8 v0 = *(const bf16x8*)(vb + (size_t)(s4 * 16 + mm) * SS + jrun0);
            const bf16x8 v1 = *(const bf16x8*)(vb + (size_t)(s4 * 16 + mm) * SS + jrun1);
            const bf16x16 bv16 = __builtin_shufflevector(v0, v1,
                0,1,2,3,4,5,6,7,8,9,10,11,12,13,14,15);
            ctx[s4] = wmma_bf16(ap, bv16, ctx[s4]);
        }
    }

    // combine the two j-halves: lanes l and l^16 hold the same query row
    rs += __shfl_xor(rs, 16, 32);

    #pragma unroll
    for (int s4 = 0; s4 < 4; s4++) {
        const float vs = vsum[bh * HD + s4 * 16 + mm];
        #pragma unroll
        for (int r = 0; r < 8; r++) {
            const float rsi = __shfl(rs, r + 8 * hh, 32);   // rowsum for query i
            const int i = i0 + r + 8 * hh;
            const float val = (ctx[s4][r] + vs) / (rsi + (float)SS);
            out[(size_t)(b * SS + i) * HID + h * HD + s4 * 16 + mm] = val;
        }
    }
}

// ---------------------------------------------------------------------------
extern "C" void kernel_launch(void* const* d_in, const int* in_sizes, int n_in,
                              void* d_out, int out_size, void* d_ws, size_t ws_size,
                              hipStream_t stream) {
    (void)in_sizes; (void)n_in; (void)out_size; (void)ws_size;
    const float* x  = (const float*)d_in[0];
    const float* Wq = (const float*)d_in[1];
    const float* bq = (const float*)d_in[2];
    const float* Wk = (const float*)d_in[3];
    const float* bk = (const float*)d_in[4];
    const float* Wv = (const float*)d_in[5];
    const float* bv = (const float*)d_in[6];
    float* out = (float*)d_out;

    const size_t qkN = (size_t)BB * NH * SS * HD;   // 3,145,728 elems (bf16)
    __bf16* qd  = (__bf16*)d_ws;
    __bf16* kd  = qd + qkN;
    __bf16* vtd = kd + qkN;
    float*  vsum = (float*)(vtd + qkN);             // 24*64 f32

    qkv_kernel<<<dim3(384, 3, 1), 256, 0, stream>>>(x, Wq, bq, Wk, bk, Wv, bv,
                                                    qd, kd, vtd);
    vsum_kernel<<<dim3(BB * NH, 1, 1), 64, 0, stream>>>(vtd, vsum);
    attn_kernel<<<dim3(SS / 128, NH, BB), 256, 0, stream>>>(qd, kd, vtd, vsum, out);
}
